// GAT_26225070309439
// MI455X (gfx1250) — matrix operand order repaired and checked by reading, not compiled, not run on hardware
//
#include <hip/hip_runtime.h>
#include <hip/hip_bf16.h>

// ---------------------------------------------------------------------------
// GAT (2-layer, PyG-style) for MI455X / gfx1250, wave32, WMMA f16->f32.
//   layer1: GATConv(128 -> 50, heads=4, concat)  => 200 features
//   layer2: GATConv(200 -> 40, heads=1)          => 40 classes, log_softmax
// GEMMs: templated K, guard-free inner loop, b128 A loads, pre-packed B frags.
// Edge phase: max / sum(+store p) / normalize / wave-per-edge scatter.
// ---------------------------------------------------------------------------

#define NNODE   50000
#define FIN     128
#define HEADS1  4
#define CPH1    50
#define CH1     200      // HEADS1*CPH1
#define FOUT    40
#define X2S     224      // padded stride of layer-2 input (200 -> mult of 32)
#define NEG_SLOPE 0.2f

typedef __attribute__((ext_vector_type(16))) _Float16 v16h;
typedef __attribute__((ext_vector_type(8)))  float    v8f;

// ---------------------------------------------------------------- utilities
__global__ void k_fill(float* __restrict__ p, float v, int n) {
    int i = blockIdx.x * blockDim.x + threadIdx.x;
    if (i < n) p[i] = v;
}

__device__ __forceinline__ void atomicMaxFloat(float* a, float v) {
    // signed compare for non-negative floats, reversed unsigned for negative.
    if (v >= 0.0f) atomicMax((int*)a, __float_as_int(v));
    else           atomicMin((unsigned int*)a, __float_as_uint(v));
}

// Pack W[Ksrc x Nsrc] (f32 row-major) into WMMA B-fragment order (f16):
//   Bp[((nt*KT + kt)*32 + lane)*16 + j]  with  K = kt*32 + (j>=8)*16 + (lane>=16)*8 + (j&7),
//   N = nt*16 + (lane&15).   (CDNA5 ISA 7.12.2 wave32 B layout.)
__global__ void k_pack_b(const float* __restrict__ W, _Float16* __restrict__ Bp,
                         int Ksrc, int Nsrc, int KT, int Ntiles) {
    int i = blockIdx.x * blockDim.x + threadIdx.x;
    int tot = Ntiles * KT * 32 * 16;
    if (i >= tot) return;
    int j    = i & 15;
    int lane = (i >> 4) & 31;
    int kt   = (i >> 9) % KT;
    int nt   = (i >> 9) / KT;
    int row = kt * 32 + (j >> 3) * 16 + (lane >> 4) * 8 + (j & 7);
    int col = nt * 16 + (lane & 15);
    Bp[i] = (row < Ksrc && col < Nsrc) ? (_Float16)W[(size_t)row * Nsrc + col]
                                       : (_Float16)0.0f;
}

// ------------------------------------------------------------------- GEMM
// C[M x Nsrc] = A[M x KPAD](f32, stride Astride, cvt f16) @ Bp(packed f16)
// 8 waves / block, one 16x16 tile per wave, K fully unrolled (KPAD/32 WMMAs).
// A 16x32 f16 frag: lane m = lo, halves = two contiguous 8-float chunks at
//   k0 + hi*8 and k0 + 16 + hi*8 (32B-aligned -> 4x global_load_b128).
template<int KPAD>
__global__ void __launch_bounds__(256)
k_gemm_f16(const float* __restrict__ A, const _Float16* __restrict__ Bp,
           float* __restrict__ C, int Astride, int Cstride, int Nsrc,
           int Mtiles, int Ntiles) {
    constexpr int KT = KPAD / 32;
    const int wave = threadIdx.x >> 5;
    const int lane = threadIdx.x & 31;
    const int tile = blockIdx.x * 8 + wave;
    if (tile >= Mtiles * Ntiles) return;           // wave-uniform exit
    const int mtile = tile / Ntiles;
    const int ntile = tile % Ntiles;
    const int hi = lane >> 4, lo = lane & 15;

    const float*    arow = A  + (size_t)(mtile * 16 + lo) * Astride + hi * 8;
    const _Float16* bp   = Bp + (size_t)(ntile * KT) * 512 + (size_t)lane * 16;

    v8f acc = {};
#pragma unroll
    for (int kt = 0; kt < KT; ++kt) {
        const float4* ap = (const float4*)(arow + kt * 32);
        float4 f0 = ap[0], f1 = ap[1];   // K-local hi*8 + 0..7
        float4 f2 = ap[4], f3 = ap[5];   // K-local 16 + hi*8 + 0..7
        v16h a;
        a[0]  = (_Float16)f0.x; a[1]  = (_Float16)f0.y;
        a[2]  = (_Float16)f0.z; a[3]  = (_Float16)f0.w;
        a[4]  = (_Float16)f1.x; a[5]  = (_Float16)f1.y;
        a[6]  = (_Float16)f1.z; a[7]  = (_Float16)f1.w;
        a[8]  = (_Float16)f2.x; a[9]  = (_Float16)f2.y;
        a[10] = (_Float16)f2.z; a[11] = (_Float16)f2.w;
        a[12] = (_Float16)f3.x; a[13] = (_Float16)f3.y;
        a[14] = (_Float16)f3.z; a[15] = (_Float16)f3.w;
        v16h b = *(const v16h*)(bp + (size_t)kt * 512);  // 32B contiguous
        acc = __builtin_amdgcn_wmma_f32_16x16x32_f16(
                  false, a, false, b, (short)0, acc, false, false);
    }

    const int n = ntile * 16 + lo;
#pragma unroll
    for (int i = 0; i < 8; ++i) {
        int m = mtile * 16 + hi * 8 + i;              // C frag: VGPR i -> M
        if (n < Nsrc) C[(size_t)m * Cstride + n] = acc[i];
    }
}

// --------------------------------------------------- per-node attention dots
__global__ void k_attn_node(const float* __restrict__ h, int hs,
                            const float* __restrict__ att_s,
                            const float* __restrict__ att_d,
                            float* __restrict__ as, float* __restrict__ ad,
                            int N, int H, int Cp) {
    int t = blockIdx.x * blockDim.x + threadIdx.x;
    if (t >= N * H) return;
    int n = t / H, hh = t % H;
    const float* hp = h + (size_t)n * hs + (size_t)hh * Cp;
    const float* ws = att_s + hh * Cp;
    const float* wd = att_d + hh * Cp;
    float ss = 0.0f, sd = 0.0f;
    for (int c = 0; c < Cp; ++c) { float v = hp[c]; ss += v * ws[c]; sd += v * wd[c]; }
    as[t] = ss; ad[t] = sd;
}

// ------------------------------------------------------------- edge passes
__global__ void k_edge_max(const int* __restrict__ src, const int* __restrict__ dst,
                           const float* __restrict__ as, const float* __restrict__ ad,
                           float* __restrict__ mx, int E, int H) {
    int t = blockIdx.x * blockDim.x + threadIdx.x;
    if (t >= E * H) return;
    int e = t / H, hh = t % H;
    float ev = as[src[e] * H + hh] + ad[dst[e] * H + hh];
    ev = ev > 0.0f ? ev : NEG_SLOPE * ev;
    atomicMaxFloat(&mx[dst[e] * H + hh], ev);
}

__global__ void k_edge_sum(const int* __restrict__ src, const int* __restrict__ dst,
                           const float* __restrict__ as, const float* __restrict__ ad,
                           const float* __restrict__ mx, float* __restrict__ sm,
                           float* __restrict__ alpha, int E, int H) {
    int t = blockIdx.x * blockDim.x + threadIdx.x;
    if (t >= E * H) return;
    int e = t / H, hh = t % H;
    int d = dst[e];
    float ev = as[src[e] * H + hh] + ad[d * H + hh];
    ev = ev > 0.0f ? ev : NEG_SLOPE * ev;
    float p = expf(ev - mx[d * H + hh]);
    alpha[t] = p;
    atomicAdd(&sm[d * H + hh], p);
}

__global__ void k_edge_norm(const int* __restrict__ dst, const float* __restrict__ sm,
                            float* __restrict__ alpha, int E, int H) {
    int t = blockIdx.x * blockDim.x + threadIdx.x;
    if (t >= E * H) return;
    alpha[t] /= (sm[dst[t / H] * H + t % H] + 1e-16f);
}

// One wave per edge: lanes sweep channels; alpha precomputed per (edge,head).
__global__ void k_edge_aggr(const int* __restrict__ src, const int* __restrict__ dst,
                            const float* __restrict__ alpha,
                            const float* __restrict__ h, int hs,
                            float* __restrict__ out, int os,
                            int E, int H, int Cp) {
    int gt = blockIdx.x * blockDim.x + threadIdx.x;
    int e = gt >> 5;
    int lane = gt & 31;
    if (e >= E) return;
    int s = src[e], d = dst[e];
    int CH = H * Cp;
    for (int c = lane; c < CH; c += 32) {
        float al = alpha[(size_t)e * H + c / Cp];
        atomicAdd(&out[(size_t)d * os + c], h[(size_t)s * hs + c] * al);
    }
}

// ------------------------------------------------------------ pointwise ops
__global__ void k_bias_elu(float* __restrict__ x, const float* __restrict__ b,
                           int n /*NNODE*CH1*/) {
    int i = blockIdx.x * blockDim.x + threadIdx.x;
    if (i >= n) return;
    int node = i / CH1, c = i % CH1;
    float* p = x + (size_t)node * X2S + c;
    float v = *p + b[c];
    *p = v > 0.0f ? v : expm1f(v);
}

__global__ void k_logsoftmax(float* __restrict__ o, const float* __restrict__ b,
                             int N, int C) {
    int n = blockIdx.x * blockDim.x + threadIdx.x;
    if (n >= N) return;
    float* p = o + (size_t)n * C;
    float m = -__builtin_inff();
    for (int c = 0; c < C; ++c) { float v = p[c] + b[c]; m = v > m ? v : m; }
    float s = 0.0f;
    for (int c = 0; c < C; ++c) s += expf(p[c] + b[c] - m);
    float ls = logf(s);
    for (int c = 0; c < C; ++c) p[c] = p[c] + b[c] - m - ls;
}

// ---------------------------------------------------------------------------
extern "C" void kernel_launch(void* const* d_in, const int* in_sizes, int n_in,
                              void* d_out, int out_size, void* d_ws, size_t ws_size,
                              hipStream_t stream) {
    const float* x    = (const float*)d_in[0];
    const int*   esrc = (const int*)d_in[1];
    const int*   edst = (const int*)d_in[2];
    const float* W1   = (const float*)d_in[3];
    const float* as1w = (const float*)d_in[4];
    const float* ad1w = (const float*)d_in[5];
    const float* b1   = (const float*)d_in[6];
    const float* W2   = (const float*)d_in[7];
    const float* as2w = (const float*)d_in[8];
    const float* ad2w = (const float*)d_in[9];
    const float* b2   = (const float*)d_in[10];
    float* out        = (float*)d_out;
    const int E = in_sizes[1];

    auto cdiv  = [](int a, int b) { return (a + b - 1) / b; };
    auto align = [](size_t v) { return (v + 255) & ~(size_t)255; };

    // ---- workspace carve-up ----
    char* w = (char*)d_ws;
    float* h1 = (float*)w;           w += align((size_t)NNODE * CH1 * 4);   // 40 MB
    float* x2 = (float*)w;           w += align((size_t)NNODE * X2S * 4);   // 44.8 MB
    float* asrc1 = (float*)w;        w += align((size_t)NNODE * HEADS1 * 4);
    float* adst1 = (float*)w;        w += align((size_t)NNODE * HEADS1 * 4);
    float* max1  = (float*)w;        w += align((size_t)NNODE * HEADS1 * 4);
    float* sum1  = (float*)w;        w += align((size_t)NNODE * HEADS1 * 4);
    float* alpha1 = (float*)w;       w += align((size_t)E * HEADS1 * 4);    // 13.6 MB
    float* asrc2 = (float*)w;        w += align((size_t)NNODE * 4);
    float* adst2 = (float*)w;        w += align((size_t)NNODE * 4);
    float* max2  = (float*)w;        w += align((size_t)NNODE * 4);
    float* sum2  = (float*)w;        w += align((size_t)NNODE * 4);
    float* alpha2 = (float*)w;       w += align((size_t)E * 4);             // 3.4 MB
    _Float16* Bp1 = (_Float16*)w;    w += align((size_t)13 * 4 * 512 * 2);  // 13 ntiles, KT=4
    _Float16* Bp2 = (_Float16*)w;    w += align((size_t)3 * 7 * 512 * 2);   // 3 ntiles,  KT=7
    float* h2 = h1;  // layer-2 pre-attn features alias h1 (dead by then)

    const int TB = 256;

    // ---- pack weights into B-fragment layout ----
    k_pack_b<<<cdiv(13 * 4 * 512, TB), TB, 0, stream>>>(W1, Bp1, FIN, CH1, 4, 13);
    k_pack_b<<<cdiv(3 * 7 * 512, TB), TB, 0, stream>>>(W2, Bp2, CH1, FOUT, 7, 3);

    // ---- init accumulators / softmax state ----
    k_fill<<<cdiv(NNODE * X2S, TB), TB, 0, stream>>>(x2, 0.0f, NNODE * X2S);
    k_fill<<<cdiv(NNODE * HEADS1, TB), TB, 0, stream>>>(max1, -__builtin_inff(), NNODE * HEADS1);
    k_fill<<<cdiv(NNODE * HEADS1, TB), TB, 0, stream>>>(sum1, 0.0f, NNODE * HEADS1);
    k_fill<<<cdiv(NNODE * FOUT, TB), TB, 0, stream>>>(out, 0.0f, NNODE * FOUT);
    k_fill<<<cdiv(NNODE, TB), TB, 0, stream>>>(max2, -__builtin_inff(), NNODE);
    k_fill<<<cdiv(NNODE, TB), TB, 0, stream>>>(sum2, 0.0f, NNODE);

    // ================= layer 1 =================
    {   // h1 = x @ W1   (M=50000, K=128, N=200 over 13 n-tiles)
        int Mt = NNODE / 16, Nt = 13;
        k_gemm_f16<128><<<cdiv(Mt * Nt, 8), 256, 0, stream>>>(
            x, Bp1, h1, FIN, CH1, CH1, Mt, Nt);
    }
    k_attn_node<<<cdiv(NNODE * HEADS1, TB), TB, 0, stream>>>(
        h1, CH1, as1w, ad1w, asrc1, adst1, NNODE, HEADS1, CPH1);
    k_edge_max<<<cdiv(E * HEADS1, TB), TB, 0, stream>>>(
        esrc, edst, asrc1, adst1, max1, E, HEADS1);
    k_edge_sum<<<cdiv(E * HEADS1, TB), TB, 0, stream>>>(
        esrc, edst, asrc1, adst1, max1, sum1, alpha1, E, HEADS1);
    k_edge_norm<<<cdiv(E * HEADS1, TB), TB, 0, stream>>>(edst, sum1, alpha1, E, HEADS1);
    k_edge_aggr<<<cdiv(E * 32, TB), TB, 0, stream>>>(
        esrc, edst, alpha1, h1, CH1, x2, X2S, E, HEADS1, CPH1);
    k_bias_elu<<<cdiv(NNODE * CH1, TB), TB, 0, stream>>>(x2, b1, NNODE * CH1);

    // ================= layer 2 =================
    {   // h2 = x2 @ W2  (K=200 zero-padded to 224 in both A and packed B)
        int Mt = NNODE / 16, Nt = 3;
        k_gemm_f16<224><<<cdiv(Mt * Nt, 8), 256, 0, stream>>>(
            x2, Bp2, h2, X2S, FOUT, FOUT, Mt, Nt);
    }
    k_attn_node<<<cdiv(NNODE, TB), TB, 0, stream>>>(
        h2, FOUT, as2w, ad2w, asrc2, adst2, NNODE, 1, FOUT);
    k_edge_max<<<cdiv(E, TB), TB, 0, stream>>>(
        esrc, edst, asrc2, adst2, max2, E, 1);
    k_edge_sum<<<cdiv(E, TB), TB, 0, stream>>>(
        esrc, edst, asrc2, adst2, max2, sum2, alpha2, E, 1);
    k_edge_norm<<<cdiv(E, TB), TB, 0, stream>>>(edst, sum2, alpha2, E, 1);
    k_edge_aggr<<<cdiv(E * 32, TB), TB, 0, stream>>>(
        esrc, edst, alpha2, h2, FOUT, out, FOUT, E, 1, FOUT);
    k_logsoftmax<<<cdiv(NNODE, TB), TB, 0, stream>>>(out, b2, NNODE, FOUT);
}
